// MultiHeadAttention_32976758898703
// MI455X (gfx1250) — matrix operand order repaired and checked by reading, not compiled, not run on hardware
//
#include <hip/hip_runtime.h>
#include <hip/hip_bf16.h>

// ---------------------------------------------------------------------------
// CDNA5 (gfx1250) MHA, f16 WMMA + fp32 accumulate + double-buffered async
// global->LDS staging (ASYNCcnt pipelining).
// Pipeline:
//   1) convert X, Wq/Wk/Wv/Wo fp32 -> f16 (one bandwidth pass, ~25MB)
//   2) Q/K/V GEMMs (f16 in, f16 out, fused bias)
//   3) flash attention (register-resident online softmax on WMMA C-layout)
//   4) output projection GEMM (f16 in, fp32 out, fused bias)
// ---------------------------------------------------------------------------

typedef __attribute__((ext_vector_type(16))) _Float16 v16h;
typedef __attribute__((ext_vector_type(8)))  _Float16 v8h;
typedef __attribute__((ext_vector_type(8)))  float    v8f;
typedef __attribute__((ext_vector_type(4)))  int      v4i;

__device__ __forceinline__ v8f wmma_f16(v16h a, v16h b, v8f c) {
  return __builtin_amdgcn_wmma_f32_16x16x32_f16(
      false, a, false, b, (short)0, c, false, false);
}

// A-matrix (16x32 f16) K index for half h: lanes 0-15 K{0-7,16-23},
// lanes 16-31 K{8-15,24-31}
__device__ __forceinline__ int kmapA(int h, int lane) {
  return h + 8 * ((h >> 3) + (lane >> 4));
}
// B-matrix (32x16 f16): same K order as A: lanes 0-15 K{0-7,16-23},
// lanes 16-31 K{8-15,24-31}
__device__ __forceinline__ int kmapB(int h, int lane) {
  return h + 8 * ((h >> 3) + (lane >> 4));
}

// 16-byte async copy global->LDS (ASYNCcnt) with synchronous fallback.
#if __has_builtin(__builtin_amdgcn_global_load_async_to_lds_b128)
#define HAVE_ASYNC_LDS 1
#endif

__device__ __forceinline__ void copy_b128_g2l(const _Float16* g, _Float16* l) {
#ifdef HAVE_ASYNC_LDS
  __builtin_amdgcn_global_load_async_to_lds_b128(
      (__attribute__((address_space(1))) v4i*)g,
      (__attribute__((address_space(3))) v4i*)l, 0, 0);
#else
  *(v8h*)l = *(const v8h*)g;
#endif
}

// Wait until at most N async instructions remain in flight.
template <int N>
__device__ __forceinline__ void wait_async_le() {
#ifdef HAVE_ASYNC_LDS
#if __has_builtin(__builtin_amdgcn_s_wait_asynccnt)
  __builtin_amdgcn_s_wait_asynccnt(N);
#else
  asm volatile("s_wait_asynccnt %0" ::"i"(N) : "memory");
#endif
#endif
}

// ---------------------------------------------------------------------------
// fp32 -> f16 conversion pass (4 elems / thread)
// ---------------------------------------------------------------------------
__global__ __launch_bounds__(256, 1)
void cvt_f32_to_f16_kernel(const float* __restrict__ src,
                           _Float16* __restrict__ dst, int n) {
  const int i = (blockIdx.x * 256 + threadIdx.x) * 4;
  if (i + 3 < n) {
    const float4 v = *(const float4*)(src + i);
    dst[i + 0] = (_Float16)v.x;
    dst[i + 1] = (_Float16)v.y;
    dst[i + 2] = (_Float16)v.z;
    dst[i + 3] = (_Float16)v.w;
  }
}

// ---------------------------------------------------------------------------
// GEMM: out[N,Dout] = X[N,K](f16) @ W[K,Dout](f16) + bias(f32).
// Block tile 128x128, K-step 64, 256 threads = 8 waves (4x2), wave tile 32x64.
// Double-buffered LDS: tile t+2 issued after tile t computes; each tile is an
// 8-instruction async batch per wave (FIFO completion -> wait asynccnt<=8).
// ---------------------------------------------------------------------------
template <typename OutT>
__global__ __launch_bounds__(256, 1)
void gemm_bias_kernel(const _Float16* __restrict__ X,
                      const _Float16* __restrict__ W,
                      const float* __restrict__ bias, OutT* __restrict__ out,
                      int N, int K, int Dout) {
  constexpr int BM = 128, BN = 128, BK = 64;
  constexpr int AST = BK + 8;   // 72 halves = 144B rows (16B aligned)
  constexpr int BST = BN + 8;   // 136 halves = 272B rows (16B aligned)
  __shared__ __attribute__((aligned(16))) _Float16 As[2][BM * AST];
  __shared__ __attribute__((aligned(16))) _Float16 Bs[2][BK * BST];

  const int tid    = threadIdx.x;
  const int lane   = tid & 31;
  const int wave   = tid >> 5;
  const int wm     = wave >> 1;     // 0..3
  const int wn     = wave & 1;      // 0..1
  const int laneLo = lane & 15;
  const int laneHi = lane >> 4;

  const int rowBase = blockIdx.y * BM;
  const int colBase = blockIdx.x * BN;

  // issue one K-tile (A: 128x64, B: 64x128 halves) = 8 async instrs per wave
  auto issue_tile = [&](int kb, int buf) {
#pragma unroll
    for (int i = 0; i < 4; ++i) {
      const int c = tid + i * 256;
      const int row = c >> 3, seg = c & 7;
      copy_b128_g2l(X + (size_t)(rowBase + row) * K + kb + seg * 8,
                    &As[buf][row * AST + seg * 8]);
    }
#pragma unroll
    for (int i = 0; i < 4; ++i) {
      const int c = tid + i * 256;
      const int row = c >> 4, seg = c & 15;
      copy_b128_g2l(W + (size_t)(kb + row) * Dout + colBase + seg * 8,
                    &Bs[buf][row * BST + seg * 8]);
    }
  };

  const v8f vzero = {0.f, 0.f, 0.f, 0.f, 0.f, 0.f, 0.f, 0.f};
  v8f acc[2][4];
  for (int mt = 0; mt < 2; ++mt)
    for (int nt = 0; nt < 4; ++nt)
      acc[mt][nt] = vzero;

  const int nT = K / BK;  // >= 2
  issue_tile(0, 0);
  issue_tile(BK, 1);

  for (int t = 0; t < nT; ++t) {
    const int buf = t & 1;
    if (t + 1 < nT) wait_async_le<8>();  // oldest tile done, next in flight
    else            wait_async_le<0>();
    __syncthreads();

    const _Float16* A = As[buf];
    const _Float16* B = Bs[buf];
#pragma unroll
    for (int dc = 0; dc < 2; ++dc) {  // BK=64 -> two K=32 WMMA steps
      v16h a[2], b[4];
#pragma unroll
      for (int mt = 0; mt < 2; ++mt) {
        const _Float16* ap = &A[(wm * 32 + mt * 16) * AST + dc * 32];
#pragma unroll
        for (int h = 0; h < 16; ++h)
          a[mt][h] = ap[laneLo * AST + kmapA(h, lane)];
      }
#pragma unroll
      for (int nt = 0; nt < 4; ++nt) {
        const _Float16* bp = &B[(dc * 32) * BST + wn * 64 + nt * 16];
#pragma unroll
        for (int h = 0; h < 16; ++h)
          b[nt][h] = bp[kmapB(h, lane) * BST + laneLo];
      }
#pragma unroll
      for (int mt = 0; mt < 2; ++mt)
#pragma unroll
        for (int nt = 0; nt < 4; ++nt)
          acc[mt][nt] = wmma_f16(a[mt], b[nt], acc[mt][nt]);
    }
    __syncthreads();  // all waves done reading buf before it is overwritten
    if (t + 2 < nT) issue_tile((t + 2) * BK, buf);
  }

  // Epilogue: C layout row = v + 8*laneHi, col = laneLo per 16x16 tile
#pragma unroll
  for (int mt = 0; mt < 2; ++mt) {
#pragma unroll
    for (int nt = 0; nt < 4; ++nt) {
#pragma unroll
      for (int v = 0; v < 8; ++v) {
        const int row = rowBase + wm * 32 + mt * 16 + v + 8 * laneHi;
        const int col = colBase + wn * 64 + nt * 16 + laneLo;
        out[(size_t)row * Dout + col] = (OutT)(acc[mt][nt][v] + bias[col]);
      }
    }
  }
}

// ---------------------------------------------------------------------------
// Flash attention per (b, h, 64-query tile). 128 threads = 4 waves; wave w
// owns query rows [16w,16w+16). Row softmax stats live in fixed
// (lane-half, vgpr-slot) pairs of the WMMA C layout -> 16-lane shfl reductions.
// K/V tiles double-buffered via async batches (4 instrs per wave per tile).
// Q fragments are hoisted into registers once (loop-invariant).
// ---------------------------------------------------------------------------
__global__ __launch_bounds__(128, 1)
void attention_kernel(const _Float16* __restrict__ Qg,
                      const _Float16* __restrict__ Kg,
                      const _Float16* __restrict__ Vg,
                      _Float16* __restrict__ Og) {
  constexpr int S = 2048, D = 1024, HD = 64;
  constexpr int QT = 64, KT = 64, ST = HD + 8;  // 72 halves = 144B rows

  __shared__ __attribute__((aligned(16))) _Float16 Qs[QT * ST];
  __shared__ __attribute__((aligned(16))) _Float16 Ks[2][KT * ST];
  __shared__ __attribute__((aligned(16))) _Float16 Vs[2][KT * ST];
  __shared__ __attribute__((aligned(16))) _Float16 Ps[4][16 * ST];

  const int qt = blockIdx.x, hh0 = blockIdx.y, bb = blockIdx.z;
  const int tid    = threadIdx.x;
  const int lane   = tid & 31;
  const int wave   = tid >> 5;
  const int laneLo = lane & 15;
  const int laneHi = lane >> 4;

  const size_t headBase = ((size_t)bb * S) * D + (size_t)hh0 * HD;

  // one K/V tile = 4 async instrs per wave (2 chunks K + 2 chunks V / thread)
  auto issue_kv = [&](int kt, int buf) {
#pragma unroll
    for (int i = 0; i < 2; ++i) {
      const int c = tid + i * 128;
      const int row = c >> 2, seg = c & 3;
      const size_t g = headBase + (size_t)(kt * KT + row) * D + seg * 8;
      copy_b128_g2l(Kg + g, &Ks[buf][row * ST + seg * 8]);
      copy_b128_g2l(Vg + g, &Vs[buf][row * ST + seg * 8]);
    }
  };

  // Q tile: 64x32 halves = 256 b128 chunks, 2 per thread (issued first)
#pragma unroll
  for (int i = 0; i < 2; ++i) {
    const int c = tid + i * 128;
    const int row = c >> 2, seg = c & 3;
    copy_b128_g2l(Qg + headBase + (size_t)(qt * QT + row) * D + seg * 8,
                  &Qs[row * ST + seg * 8]);
  }
  issue_kv(0, 0);
  issue_kv(1, 1);

  // Hoist loop-invariant Q A-fragments: FIFO async retirement means
  // wait<=8 (kv0+kv1 still in flight) guarantees the older Q batch landed.
  wait_async_le<8>();
  __syncthreads();
  v16h aq[2];
#pragma unroll
  for (int dc = 0; dc < 2; ++dc) {
    const _Float16* ap = &Qs[wave * 16 * ST + dc * 32];
#pragma unroll
    for (int h = 0; h < 16; ++h) aq[dc][h] = ap[laneLo * ST + kmapA(h, lane)];
  }

  const v8f vzero = {0.f, 0.f, 0.f, 0.f, 0.f, 0.f, 0.f, 0.f};
  float m_i[8], l_i[8];
  v8f o_acc[4];
#pragma unroll
  for (int v = 0; v < 8; ++v) { m_i[v] = -1e30f; l_i[v] = 0.f; }
#pragma unroll
  for (int ot = 0; ot < 4; ++ot) o_acc[ot] = vzero;

  constexpr int nT = S / KT;  // 32
  for (int kt = 0; kt < nT; ++kt) {
    const int buf = kt & 1;
    if (kt + 1 < nT) wait_async_le<4>();  // this tile complete, next in flight
    else             wait_async_le<0>();
    __syncthreads();

    // S = Q * K^T (16 rows x 64 keys per wave, fp32 accum)
    v8f s_acc[4] = {vzero, vzero, vzero, vzero};
#pragma unroll
    for (int dc = 0; dc < 2; ++dc) {
#pragma unroll
      for (int nt = 0; nt < 4; ++nt) {
        v16h bk;  // B[d][n] = K[n][d]
        const _Float16* bp = &Ks[buf][nt * 16 * ST + dc * 32];
#pragma unroll
        for (int h = 0; h < 16; ++h) bk[h] = bp[laneLo * ST + kmapB(h, lane)];
        s_acc[nt] = wmma_f16(aq[dc], bk, s_acc[nt]);
      }
    }

    constexpr float scale = 0.125f;  // 1/sqrt(HD)
#pragma unroll
    for (int v = 0; v < 8; ++v) {
      float rmax = -1e30f;
#pragma unroll
      for (int nt = 0; nt < 4; ++nt) {
        s_acc[nt][v] *= scale;
        rmax = fmaxf(rmax, s_acc[nt][v]);
      }
#pragma unroll
      for (int m = 1; m < 16; m <<= 1)
        rmax = fmaxf(rmax, __shfl_xor(rmax, m, 32));
      const float mnew = fmaxf(m_i[v], rmax);
      const float corr = __expf(m_i[v] - mnew);
      float rsum = 0.f;
#pragma unroll
      for (int nt = 0; nt < 4; ++nt) {
        const float p = __expf(s_acc[nt][v] - mnew);
        s_acc[nt][v] = p;
        rsum += p;
      }
#pragma unroll
      for (int m = 1; m < 16; m <<= 1) rsum += __shfl_xor(rsum, m, 32);
      m_i[v] = mnew;
      l_i[v] = l_i[v] * corr + rsum;
#pragma unroll
      for (int ot = 0; ot < 4; ++ot) o_acc[ot][v] *= corr;
    }

    // P: C-layout -> A-layout via LDS
#pragma unroll
    for (int nt = 0; nt < 4; ++nt)
#pragma unroll
      for (int v = 0; v < 8; ++v)
        Ps[wave][(v + 8 * laneHi) * ST + nt * 16 + laneLo] =
            (_Float16)s_acc[nt][v];
    __syncthreads();

    // O += P * V
#pragma unroll
    for (int kc = 0; kc < 2; ++kc) {
      v16h apf;
      const _Float16* pp = &Ps[wave][kc * 32];
#pragma unroll
      for (int h = 0; h < 16; ++h) apf[h] = pp[laneLo * ST + kmapA(h, lane)];
#pragma unroll
      for (int ot = 0; ot < 4; ++ot) {
        v16h bv;
        const _Float16* bp = &Vs[buf][kc * 32 * ST + ot * 16];
#pragma unroll
        for (int h = 0; h < 16; ++h) bv[h] = bp[kmapB(h, lane) * ST + laneLo];
        o_acc[ot] = wmma_f16(apf, bv, o_acc[ot]);
      }
    }

    __syncthreads();  // all waves done reading Ks/Vs[buf] before overwrite
    if (kt + 2 < nT) issue_kv(kt + 2, buf);
  }

  // finalize: O /= l, store f16 [b, s, h, hd]
#pragma unroll
  for (int v = 0; v < 8; ++v) {
    const float inv = 1.f / l_i[v];
    const int row = qt * QT + wave * 16 + v + 8 * laneHi;
#pragma unroll
    for (int ot = 0; ot < 4; ++ot) {
      const int col = ot * 16 + laneLo;
      Og[headBase + (size_t)row * D + col] = (_Float16)(o_acc[ot][v] * inv);
    }
  }
}

// ---------------------------------------------------------------------------
extern "C" void kernel_launch(void* const* d_in, const int* in_sizes, int n_in,
                              void* d_out, int out_size, void* d_ws,
                              size_t ws_size, hipStream_t stream) {
  (void)n_in; (void)out_size; (void)ws_size;
  const float* X  = (const float*)d_in[0];
  const float* wq = (const float*)d_in[1];
  const float* bq = (const float*)d_in[2];
  const float* wk = (const float*)d_in[3];
  const float* bk = (const float*)d_in[4];
  const float* wv = (const float*)d_in[5];
  const float* bv = (const float*)d_in[6];
  const float* wo = (const float*)d_in[7];
  const float* bo = (const float*)d_in[8];
  float* out = (float*)d_out;

  const int D = 1024;
  const int N = in_sizes[0] / D;           // b*s = 4096
  const size_t xe = (size_t)N * D;         // 4M elems
  const size_t we = (size_t)D * D;         // 1M elems

  // f16 workspace layout (halves): Xh | Wq | Wk | Wv | Wo | Q | K | V | O
  _Float16* Xh  = (_Float16*)d_ws;
  _Float16* Wqh = Xh + xe;
  _Float16* Wkh = Wqh + we;
  _Float16* Wvh = Wkh + we;
  _Float16* Woh = Wvh + we;
  _Float16* Qh  = Woh + we;
  _Float16* Kh  = Qh + xe;
  _Float16* Vh  = Kh + xe;
  _Float16* Oh  = Vh + xe;                 // total 48 MB

  // 1) fp32 -> f16 conversion passes
  cvt_f32_to_f16_kernel<<<(int)(xe / 1024), 256, 0, stream>>>(X, Xh, (int)xe);
  cvt_f32_to_f16_kernel<<<(int)(we / 1024), 256, 0, stream>>>(wq, Wqh, (int)we);
  cvt_f32_to_f16_kernel<<<(int)(we / 1024), 256, 0, stream>>>(wk, Wkh, (int)we);
  cvt_f32_to_f16_kernel<<<(int)(we / 1024), 256, 0, stream>>>(wv, Wvh, (int)we);
  cvt_f32_to_f16_kernel<<<(int)(we / 1024), 256, 0, stream>>>(wo, Woh, (int)we);

  // 2) Q/K/V projections
  dim3 gGemm(D / 128, N / 128);
  gemm_bias_kernel<_Float16><<<gGemm, 256, 0, stream>>>(Xh, Wqh, bq, Qh, N, D, D);
  gemm_bias_kernel<_Float16><<<gGemm, 256, 0, stream>>>(Xh, Wkh, bk, Kh, N, D, D);
  gemm_bias_kernel<_Float16><<<gGemm, 256, 0, stream>>>(Xh, Wvh, bv, Vh, N, D, D);

  // 3) attention
  dim3 gAttn(2048 / 64, 16, N / 2048);
  attention_kernel<<<gAttn, 128, 0, stream>>>(Qh, Kh, Vh, Oh);

  // 4) output projection (fp32 out)
  gemm_bias_kernel<float><<<gGemm, 256, 0, stream>>>(Oh, Woh, bo, out, N, D, D);
}
